// Block_5454608466019
// MI455X (gfx1250) — matrix operand order repaired and checked by reading, not compiled, not run on hardware
//
#include <hip/hip_runtime.h>
#include <stdint.h>

// Problem dims (from reference setup_inputs; expert_capacity == 2048 == CAP_)
#define G_   8
#define S_   2048
#define D_   1024
#define H_   4096
#define E_   8
#define T_   (G_ * S_)   // 16384
#define CAP_ 2048

typedef __attribute__((ext_vector_type(16))) __bf16 v16bf;
typedef __attribute__((ext_vector_type(8)))  float  v8f;

// ---- gfx1250 async global->LDS DMA (ASYNCcnt-tracked), if toolchain exposes it
#if defined(__has_builtin)
#  if __has_builtin(__builtin_amdgcn_global_load_async_to_lds_b128) && \
      __has_builtin(__builtin_amdgcn_s_wait_asynccnt)
#    define HAVE_ASYNC_LDS 1
#  endif
#endif
#ifndef HAVE_ASYNC_LDS
#  define HAVE_ASYNC_LDS 0
#endif

#if HAVE_ASYNC_LDS
typedef int v4i __attribute__((vector_size(16)));
typedef __attribute__((address_space(1))) v4i* g4p;  // global ptr to 16B vector
typedef __attribute__((address_space(3))) v4i* l4p;  // LDS ptr to 16B vector
__device__ __forceinline__ void async_cp16(void* lds, const void* g) {
  __builtin_amdgcn_global_load_async_to_lds_b128(
      (g4p)(void*)(g), (l4p)(lds), /*offset=*/0, /*cpol=*/0);
}
#endif

__device__ __forceinline__ unsigned short f2bf(float f) {
  union { float f; unsigned int u; } c; c.f = f;
  unsigned int u = c.u;
  unsigned int r = u + 0x7FFFu + ((u >> 16) & 1u);   // round-to-nearest-even
  return (unsigned short)(r >> 16);
}

__device__ __forceinline__ float gelu_tanh(float x) {
  const float k0 = 0.7978845608028654f;  // sqrt(2/pi)
  const float k1 = 0.044715f;
  float x3 = x * x * x;
  return 0.5f * x * (1.0f + tanhf(k0 * (x + k1 * x3)));
}

// ---------------- utility kernels ----------------

__global__ void zero_f32(float* __restrict__ p, long long n) {
  long long i = (long long)blockIdx.x * blockDim.x + threadIdx.x;
  long long stride = (long long)gridDim.x * blockDim.x;
  for (; i < n; i += stride) p[i] = 0.0f;
}

__global__ void cvt_f32_bf16(const float* __restrict__ in,
                             unsigned short* __restrict__ out, long long n) {
  long long i = (long long)blockIdx.x * blockDim.x + threadIdx.x;
  long long stride = (long long)gridDim.x * blockDim.x;
  for (; i < n; i += stride) out[i] = f2bf(in[i]);
}

// Batched f32 (B,K,N) -> bf16 (B,N,K) transpose-convert (32x32 LDS tiles).
__global__ void cvt_transpose_bf16(const float* __restrict__ in,
                                   unsigned short* __restrict__ out,
                                   int K, int N) {
  long long b = blockIdx.z;
  const float* inb = in + b * (long long)K * N;
  unsigned short* outb = out + b * (long long)N * K;
  __shared__ float tile[32][33];
  int n0 = blockIdx.x * 32, k0 = blockIdx.y * 32;
  int tx = threadIdx.x & 31, ty = threadIdx.x >> 5;  // 256 thr: ty 0..7
#pragma unroll
  for (int r = ty; r < 32; r += 8)
    tile[r][tx] = inb[(long long)(k0 + r) * N + n0 + tx];
  __syncthreads();
#pragma unroll
  for (int r = ty; r < 32; r += 8)
    outb[(long long)(n0 + r) * K + k0 + tx] = f2bf(tile[tx][r]);
}

// ---------------- router: logits + softmax, write probs (E,T) ----------------

__global__ void router_kernel(const float* __restrict__ x,
                              const float* __restrict__ gate_w, // (D,E)
                              float* __restrict__ probs) {      // (E,T)
  int t = blockIdx.x;
  int tid = threadIdx.x;  // 128
  const float* xr = x + (long long)t * D_;
  float acc[E_];
#pragma unroll
  for (int e = 0; e < E_; ++e) acc[e] = 0.0f;
  for (int d = tid; d < D_; d += 128) {
    float xv = xr[d];
    const float* gw = gate_w + (long long)d * E_;
#pragma unroll
    for (int e = 0; e < E_; ++e) acc[e] += xv * gw[e];
  }
  __shared__ float red[E_ * 128];
#pragma unroll
  for (int e = 0; e < E_; ++e) red[e * 128 + tid] = acc[e];
  __syncthreads();
  for (int off = 64; off > 0; off >>= 1) {
    if (tid < off) {
#pragma unroll
      for (int e = 0; e < E_; ++e) red[e * 128 + tid] += red[e * 128 + tid + off];
    }
    __syncthreads();
  }
  if (tid == 0) {
    float mx = -1e30f;
#pragma unroll
    for (int e = 0; e < E_; ++e) mx = fmaxf(mx, red[e * 128]);
    float ex[E_], s = 0.0f;
#pragma unroll
    for (int e = 0; e < E_; ++e) { ex[e] = expf(red[e * 128] - mx); s += ex[e]; }
    float inv = 1.0f / s;
#pragma unroll
    for (int e = 0; e < E_; ++e) probs[(long long)e * T_ + t] = ex[e] * inv;
  }
}

// ---------------- per-expert top-k (k = CAP_) via threshold bisection ----------------

__global__ void topk_kernel(const float* __restrict__ probs, // (E,T)
                            int* __restrict__ idxOut,        // (E,CAP)
                            float* __restrict__ scoreOut) {  // (E,CAP)
  int e = blockIdx.x;
  int tid = threadIdx.x;  // 256
  const float* p = probs + (long long)e * T_;
  __shared__ int redc[256];
  __shared__ int s_cnt;

  float lo = 0.0f, hi = 1.0f;
  for (int it = 0; it < 24; ++it) {
    float mid = 0.5f * (lo + hi);
    int c = 0;
    for (int i = tid; i < T_; i += 256) c += (p[i] > mid) ? 1 : 0;
    redc[tid] = c;
    __syncthreads();
    for (int off = 128; off > 0; off >>= 1) {
      if (tid < off) redc[tid] += redc[tid + off];
      __syncthreads();
    }
    int total = redc[0];
    __syncthreads();
    if (total > CAP_) lo = mid; else hi = mid;
  }
  if (tid == 0) s_cnt = 0;
  __syncthreads();
  for (int i = tid; i < T_; i += 256) {
    float v = p[i];
    if (v > hi) {
      int pos = atomicAdd(&s_cnt, 1);
      if (pos < CAP_) { idxOut[e * CAP_ + pos] = i; scoreOut[e * CAP_ + pos] = v; }
    }
  }
  __syncthreads();
  for (int i = tid; i < T_; i += 256) {
    float v = p[i];
    if (v <= hi && v > lo) {
      int pos = atomicAdd(&s_cnt, 1);
      if (pos < CAP_) { idxOut[e * CAP_ + pos] = i; scoreOut[e * CAP_ + pos] = v; }
    }
  }
  __syncthreads();
  int start = s_cnt < CAP_ ? s_cnt : CAP_;
  for (int pos = start + tid; pos < CAP_; pos += 256) {
    idxOut[e * CAP_ + pos] = 0;
    scoreOut[e * CAP_ + pos] = 0.0f;
  }
}

// ---------------- bf16 WMMA GEMM core ----------------
// C[M,N] = act(A[M,K] @ Wt[N,K]^T + bias); tile 128x128x32, 8 waves,
// wave = 64x32 = 4(M) x 2(N) v_wmma_f32_16x16x32_bf16 tiles.
// Both LDS tiles are (row, k) with 64B rows -> all staging is contiguous 16B,
// fragments are two ds_load_b128 each per the ISA 16-bit A/B layouts.

__device__ __forceinline__ void wmma_step(const unsigned short* __restrict__ bufA,
                                          const unsigned short* __restrict__ bufB,
                                          int wm, int wn, int half, int lr,
                                          v8f acc[4][2]) {
  union Frag { v16bf v; float4 f[2]; };
  Frag fa[4], fb[2];
#pragma unroll
  for (int mt = 0; mt < 4; ++mt) {
    int rowA = wm * 64 + mt * 16 + lr;
    const char* base = (const char*)bufA + rowA * 64;
    fa[mt].f[0] = *(const float4*)(base + half * 16);       // K 0..7 / 8..15
    fa[mt].f[1] = *(const float4*)(base + 32 + half * 16);  // K 16..23 / 24..31
  }
#pragma unroll
  for (int nt = 0; nt < 2; ++nt) {
    int col = wn * 32 + nt * 16 + lr;
    const char* base = (const char*)bufB + col * 64;
    fb[nt].f[0] = *(const float4*)(base + half * 16);       // K 0..7 / 8..15
    fb[nt].f[1] = *(const float4*)(base + 32 + half * 16);  // K 16..23 / 24..31
  }
#pragma unroll
  for (int mt = 0; mt < 4; ++mt)
#pragma unroll
    for (int nt = 0; nt < 2; ++nt)
      acc[mt][nt] = __builtin_amdgcn_wmma_f32_16x16x32_bf16(
          false, fa[mt].v, false, fb[nt].v, (short)0, acc[mt][nt], false, false);
}

template <bool SCATTER>
__global__ void __launch_bounds__(256)
gemm_bf16(const unsigned short* __restrict__ A, int ldA,  // bf16 (rows x ldA)
          const unsigned short* __restrict__ Wt,          // bf16 (N x K) pre-transposed
          const float* __restrict__ bias,                 // (N)
          const int* __restrict__ rowIdx,                 // may be null
          int gatherA, int rowBaseA, int rowBaseOut,
          const float* __restrict__ scale,                // may be null
          float* __restrict__ outF, unsigned short* __restrict__ outH,
          int M, int N, int K) {
  __shared__ unsigned short ldsA[2][128 * 32];
  __shared__ unsigned short ldsB[2][128 * 32];

  int tid  = threadIdx.x;
  int lane = tid & 31;
  int wid  = tid >> 5;
  int wm   = wid & 1;
  int wn   = wid >> 1;
  int half = lane >> 4;
  int lr   = lane & 15;
  int mBase = blockIdx.y * 128;
  int nBase = blockIdx.x * 128;

  v8f acc[4][2];
#pragma unroll
  for (int mt = 0; mt < 4; ++mt)
#pragma unroll
    for (int nt = 0; nt < 2; ++nt)
#pragma unroll
      for (int r = 0; r < 8; ++r) acc[mt][nt][r] = 0.0f;

  // Staging geometry: thread copies rows r0 and r0+64, 16B chunk cc (of 4).
  int r0 = tid >> 2, cc = tid & 3;
  int r1 = r0 + 64;
  long long arow0 = gatherA ? (long long)rowIdx[mBase + r0]
                            : (long long)(rowBaseA + mBase + r0);
  long long arow1 = gatherA ? (long long)rowIdx[mBase + r1]
                            : (long long)(rowBaseA + mBase + r1);
  const unsigned short* aSrc0 = A + arow0 * ldA + cc * 8;
  const unsigned short* aSrc1 = A + arow1 * ldA + cc * 8;
  const unsigned short* bSrc0 = Wt + (long long)(nBase + r0) * K + cc * 8;
  const unsigned short* bSrc1 = Wt + (long long)(nBase + r1) * K + cc * 8;
  int aOff0 = r0 * 32 + cc * 8;
  int aOff1 = r1 * 32 + cc * 8;

  int steps = K >> 5;

#if HAVE_ASYNC_LDS
  // ---- async DMA path: LDS written directly, ASYNCcnt-tracked, double-buffered
  async_cp16(&ldsA[0][aOff0], aSrc0);
  async_cp16(&ldsA[0][aOff1], aSrc1);
  async_cp16(&ldsB[0][aOff0], bSrc0);
  async_cp16(&ldsB[0][aOff1], bSrc1);
  for (int s = 0; s < steps; ++s) {
    int cur = s & 1;
    __builtin_amdgcn_s_wait_asynccnt(0);  // my copies into buf[cur] are done
    __syncthreads();                      // everyone's copies visible / reads drained
    if (s + 1 < steps) {
      int k = (s + 1) << 5;
      int nb = cur ^ 1;
      async_cp16(&ldsA[nb][aOff0], aSrc0 + k);
      async_cp16(&ldsA[nb][aOff1], aSrc1 + k);
      async_cp16(&ldsB[nb][aOff0], bSrc0 + k);
      async_cp16(&ldsB[nb][aOff1], bSrc1 + k);
    }
    wmma_step(&ldsA[cur][0], &ldsB[cur][0], wm, wn, half, lr, acc);
  }
#else
  // ---- fallback: register-staged, double-buffered
  float4 ra0 = *(const float4*)aSrc0;
  float4 ra1 = *(const float4*)aSrc1;
  float4 rb0 = *(const float4*)bSrc0;
  float4 rb1 = *(const float4*)bSrc1;
  for (int s = 0; s < steps; ++s) {
    int cur = s & 1;
    *(float4*)(&ldsA[cur][aOff0]) = ra0;
    *(float4*)(&ldsA[cur][aOff1]) = ra1;
    *(float4*)(&ldsB[cur][aOff0]) = rb0;
    *(float4*)(&ldsB[cur][aOff1]) = rb1;
    if (s + 1 < steps) {
      int k = (s + 1) << 5;
      ra0 = *(const float4*)(aSrc0 + k);
      ra1 = *(const float4*)(aSrc1 + k);
      rb0 = *(const float4*)(bSrc0 + k);
      rb1 = *(const float4*)(bSrc1 + k);
    }
    __syncthreads();
    wmma_step(&ldsA[cur][0], &ldsB[cur][0], wm, wn, half, lr, acc);
  }
#endif

  // ---- epilogue; C layout: VGPR r -> M = r + 8*half, N = lr
#pragma unroll
  for (int mt = 0; mt < 4; ++mt) {
#pragma unroll
    for (int nt = 0; nt < 2; ++nt) {
      int ncol = nBase + wn * 32 + nt * 16 + lr;
      float b = bias[ncol];
#pragma unroll
      for (int r = 0; r < 8; ++r) {
        int mrow = mBase + wm * 64 + mt * 16 + half * 8 + r;
        float v = acc[mt][nt][r] + b;
        if (SCATTER) {
          if (scale) v *= scale[mrow];
          int tok = rowIdx ? rowIdx[mrow] : (rowBaseOut + mrow);
          atomicAdd(&outF[(long long)tok * N + ncol], v);
        } else {
          outH[(long long)mrow * N + ncol] = f2bf(gelu_tanh(v));
        }
      }
    }
  }
}

// ---------------- host orchestration ----------------

extern "C" void kernel_launch(void* const* d_in, const int* in_sizes, int n_in,
                              void* d_out, int out_size, void* d_ws, size_t ws_size,
                              hipStream_t stream) {
  const float* x      = (const float*)d_in[0];
  const float* gate_w = (const float*)d_in[1];
  const float* ew1    = (const float*)d_in[2];
  const float* eb1    = (const float*)d_in[3];
  const float* ew2    = (const float*)d_in[4];
  const float* eb2    = (const float*)d_in[5];
  const float* sw1    = (const float*)d_in[6];
  const float* sb1    = (const float*)d_in[7];
  const float* sw2    = (const float*)d_in[8];
  const float* sb2    = (const float*)d_in[9];
  float* out = (float*)d_out;

  char* ws = (char*)d_ws;
  size_t off = 0;
  auto alloc = [&](size_t bytes) -> char* {
    char* p = ws + off;
    off = (off + bytes + 255) & ~(size_t)255;
    return p;
  };
  unsigned short* xb    = (unsigned short*)alloc((size_t)T_ * D_ * 2);
  unsigned short* ew1t  = (unsigned short*)alloc((size_t)E_ * D_ * H_ * 2);  // (E,H,D)
  unsigned short* ew2t  = (unsigned short*)alloc((size_t)E_ * H_ * D_ * 2);  // (E,D,H)
  unsigned short* sw1t  = (unsigned short*)alloc((size_t)D_ * H_ * 2);       // (H,D)
  unsigned short* sw2t  = (unsigned short*)alloc((size_t)H_ * D_ * 2);       // (D,H)
  float*          probs = (float*)alloc((size_t)E_ * T_ * 4);
  int*            idx   = (int*)alloc((size_t)E_ * CAP_ * 4);
  float*          score = (float*)alloc((size_t)E_ * CAP_ * 4);
  unsigned short* h     = (unsigned short*)alloc((size_t)CAP_ * H_ * 2);
  (void)in_sizes; (void)n_in; (void)out_size; (void)ws_size;

  zero_f32<<<4096, 256, 0, stream>>>(out, (long long)T_ * D_);

  // one-time bf16 conversions; weights also transposed to (N,K) for the GEMM B-path
  cvt_f32_bf16<<<2048, 256, 0, stream>>>(x, xb, (long long)T_ * D_);
  cvt_transpose_bf16<<<dim3(H_ / 32, D_ / 32, E_), 256, 0, stream>>>(ew1, ew1t, D_, H_);
  cvt_transpose_bf16<<<dim3(D_ / 32, H_ / 32, E_), 256, 0, stream>>>(ew2, ew2t, H_, D_);
  cvt_transpose_bf16<<<dim3(H_ / 32, D_ / 32, 1), 256, 0, stream>>>(sw1, sw1t, D_, H_);
  cvt_transpose_bf16<<<dim3(D_ / 32, H_ / 32, 1), 256, 0, stream>>>(sw2, sw2t, H_, D_);

  router_kernel<<<T_, 128, 0, stream>>>(x, gate_w, probs);
  topk_kernel<<<E_, 256, 0, stream>>>(probs, idx, score);

  dim3 blk(256);
  for (int e = 0; e < E_; ++e) {
    gemm_bf16<false><<<dim3(H_ / 128, CAP_ / 128), blk, 0, stream>>>(
        xb, D_, ew1t + (size_t)e * H_ * D_, eb1 + (size_t)e * H_,
        idx + (size_t)e * CAP_, /*gatherA=*/1, 0, 0,
        nullptr, nullptr, h, CAP_, H_, D_);
    gemm_bf16<true><<<dim3(D_ / 128, CAP_ / 128), blk, 0, stream>>>(
        h, H_, ew2t + (size_t)e * D_ * H_, eb2 + (size_t)e * D_,
        idx + (size_t)e * CAP_, /*gatherA=*/0, 0, 0,
        score + (size_t)e * CAP_, out, nullptr, CAP_, D_, H_);
  }
  for (int g = 0; g < G_; ++g) {
    gemm_bf16<false><<<dim3(H_ / 128, S_ / 128), blk, 0, stream>>>(
        xb, D_, sw1t, sb1,
        nullptr, /*gatherA=*/0, g * S_, 0,
        nullptr, nullptr, h, S_, H_, D_);
    gemm_bf16<true><<<dim3(D_ / 128, S_ / 128), blk, 0, stream>>>(
        h, H_, sw2t, sb2,
        nullptr, /*gatherA=*/0, 0, g * S_,
        nullptr, out, nullptr, S_, D_, H_);
  }
}
